// TopoAggregator_33217277067466
// MI455X (gfx1250) — matrix operand order repaired and checked by reading, not compiled, not run on hardware
//
#include <hip/hip_runtime.h>
#include <hip/hip_bf16.h>

// ---------------- CDNA5 WMMA vector types ----------------
typedef __attribute__((ext_vector_type(16))) __bf16 v16bf;
typedef __attribute__((ext_vector_type(8)))  __bf16 v8bf;
typedef __attribute__((ext_vector_type(8)))  float  v8f;

// Monotonic uint encoding of float for atomicMax-based segment max
__device__ __forceinline__ unsigned fenc(float f) {
    unsigned b = __float_as_uint(f);
    return (b & 0x80000000u) ? ~b : (b | 0x80000000u);
}
__device__ __forceinline__ float fdec(unsigned u) {
    unsigned b = (u & 0x80000000u) ? (u & 0x7FFFFFFFu) : ~u;
    return __uint_as_float(b);
}

// ---------------- Kernel 0: embedding gather -> bf16 h0 ----------------
__global__ void embed_kernel(const float* __restrict__ nodes,
                             const float* __restrict__ temb,
                             const float* __restrict__ tyemb,
                             const float* __restrict__ bwemb,
                             const float* __restrict__ ilemb,
                             __bf16* __restrict__ h0, int N) {
    int gid = blockIdx.x * blockDim.x + threadIdx.x;
    if (gid >= N * 64) return;
    int n = gid >> 6, f = gid & 63;
    const float* row = nodes + (size_t)n * 20;
    float v;
    if (f < 16)      v = temb[(int)row[0] * 16 + f];
    else if (f < 32) v = tyemb[(int)row[1] * 16 + (f - 16)];
    else if (f < 40) v = bwemb[(int)row[2] * 8 + (f - 32)];
    else if (f < 48) v = ilemb[(int)row[3] * 8 + (f - 40)];
    else             v = row[4 + (f - 48)];
    h0[gid] = (__bf16)v;
}

// ---------------- relation mean (sum reduction) ----------------
__global__ void zero16_kernel(float* p) { if (threadIdx.x < 16 && blockIdx.x == 0) p[threadIdx.x] = 0.f; }

__global__ void relsum_kernel(const float* __restrict__ relation, int E, float* __restrict__ rel_sum) {
    int tid = blockIdx.x * blockDim.x + threadIdx.x;
    int k = tid & 15;
    int r0 = tid >> 4;
    int stride = (gridDim.x * blockDim.x) >> 4;
    float acc = 0.f;
    for (int r = r0; r < E; r += stride) acc += relation[(size_t)r * 16 + k];
    unsafeAtomicAdd(rel_sum + k, acc);
}

// ep_self[j] = mean_rel @ we[j][:] ; j in [0,256)
__global__ void epself_kernel(const float* __restrict__ rel_sum, float invE,
                              const float* __restrict__ we_layer, float* __restrict__ ep_self) {
    int j = threadIdx.x;
    float acc = 0.f;
#pragma unroll
    for (int k = 0; k < 16; ++k) acc += (rel_sum[k] * invE) * we_layer[j * 16 + k];
    ep_self[j] = acc;
}

// ---------------- Pack W (Nout x 64 f32) into per-lane WMMA B-fragment layout ----------------
// Bpack[((kstep*NT + nt)*32 + lane)*16 + e] = bf16( W[(nt*16 + (lane&15))*64 + kstep*32 + (lane>>4)*8 + (e&7) + (e>>3)*16] )
template <int NT>
__global__ void pack_w_kernel(const float* __restrict__ W, __bf16* __restrict__ Bpack) {
    int idx = blockIdx.x * blockDim.x + threadIdx.x;
    if (idx >= 1024 * NT) return;
    int e     = idx & 15;
    int lane  = (idx >> 4) & 31;
    int t     = idx >> 9;
    int nt    = t % NT;
    int kstep = t / NT;
    int n = nt * 16 + (lane & 15);
    int k = kstep * 32 + (lane >> 4) * 8 + (e & 7) + (e >> 3) * 16;
    Bpack[idx] = (__bf16)W[(size_t)n * 64 + k];
}

// ---------------- WMMA GEMM: out[M x NT*16] = A(M x 64, bf16) @ W^T + bias ----------------
// One wave computes a 64 x (NT*16) tile: A fragments for 4 M-tiles held in registers,
// each B fragment load (pre-packed bf16, one 32B v16bf) feeds 4 WMMAs.
template <int NT>
__global__ void gemm_wmma_kernel(const __bf16* __restrict__ A,
                                 const __bf16* __restrict__ Bpack,
                                 const float* __restrict__ bias,
                                 float* __restrict__ outF,
                                 __bf16* __restrict__ outB,
                                 int M) {
    const int lane  = threadIdx.x & 31;
    const int wave  = threadIdx.x >> 5;
    const int mbase = blockIdx.x * 512 + wave * 64;   // 4 M-tiles of 16 per wave
    if (mbase >= M) return;

    const int kbase = (lane >> 4) * 8;                // A: K chunk base
    const int ncol  = lane & 15;                      // B/C/D column
    const int rbase = (lane >> 4) * 8;                // C/D row base

    // Load A fragments: a[mt][kstep]
    v16bf a[4][2];
#pragma unroll
    for (int mt = 0; mt < 4; ++mt) {
        int rowA = mbase + mt * 16 + (lane & 15);
        int rA = rowA < M ? rowA : M - 1;             // clamp; stores guarded
        const __bf16* arow = A + (size_t)rA * 64;
#pragma unroll
        for (int kstep = 0; kstep < 2; ++kstep) {
            const __bf16* ap = arow + kstep * 32 + kbase;
            v8bf c0 = *(const v8bf*)(ap);
            v8bf c1 = *(const v8bf*)(ap + 16);
#pragma unroll
            for (int i = 0; i < 8; ++i) { a[mt][kstep][i] = c0[i]; a[mt][kstep][8 + i] = c1[i]; }
        }
    }

#pragma unroll
    for (int nt = 0; nt < NT; ++nt) {
        v8f acc[4];
#pragma unroll
        for (int mt = 0; mt < 4; ++mt)
#pragma unroll
            for (int i = 0; i < 8; ++i) acc[mt][i] = 0.f;

#pragma unroll
        for (int kstep = 0; kstep < 2; ++kstep) {
            v16bf b = *(const v16bf*)(Bpack + ((size_t)(kstep * NT + nt) * 32 + lane) * 16);
#pragma unroll
            for (int mt = 0; mt < 4; ++mt) {
                acc[mt] = __builtin_amdgcn_wmma_f32_16x16x32_bf16(
                    false, a[mt][kstep], false, b, (short)0, acc[mt], false, false);
            }
        }

        const int n = nt * 16 + ncol;
        const float bv = bias[n];
#pragma unroll
        for (int mt = 0; mt < 4; ++mt) {
#pragma unroll
            for (int r = 0; r < 8; ++r) {
                int m = mbase + mt * 16 + r + rbase;
                if (m < M) {
                    float v = acc[mt][r] + bv;
                    size_t off = (size_t)m * (NT * 16) + n;
                    if (outF) outF[off] = v;
                    if (outB) outB[off] = (__bf16)v;
                }
            }
        }
    }
}

// ---------------- per-layer accumulator init ----------------
__global__ void layer_init_kernel(float* __restrict__ outacc, unsigned* __restrict__ mx,
                                  float* __restrict__ den, int N) {
    size_t gid = (size_t)blockIdx.x * blockDim.x + threadIdx.x;
    if (gid < (size_t)N * 256) outacc[gid] = 0.f;
    if (gid < (size_t)N * 4) { mx[gid] = 0u; den[gid] = 0.f; }
}

// ---------------- Pass 1: edge logits + segment max ----------------
__global__ void edge_logits_kernel(const __bf16* __restrict__ xl, const __bf16* __restrict__ xr,
                                   const float* __restrict__ relation,
                                   const float* __restrict__ we_layer,   // 256 x 16
                                   const float* __restrict__ att_layer,  // 4 x 64
                                   const float* __restrict__ ep_self,    // 256
                                   const int* __restrict__ edges, int E, int Et,
                                   float* __restrict__ logits, unsigned* __restrict__ mx) {
    __shared__ float we_s[4096];
    __shared__ float att_s[256];
    for (int i = threadIdx.x; i < 4096; i += blockDim.x) we_s[i] = we_layer[i];
    for (int i = threadIdx.x; i < 256; i += blockDim.x) att_s[i] = att_layer[i];
    __syncthreads();

    int gid = blockIdx.x * blockDim.x + threadIdx.x;
    if (gid >= Et * 4) return;
    int e = gid >> 2, h = gid & 3;
    int src, dst;
    bool real = e < E;
    if (real) { src = edges[e]; dst = edges[E + e]; }
    else      { src = dst = e - E; }

    float rel[16];
    if (real) {
#pragma unroll
        for (int k = 0; k < 16; ++k) rel[k] = relation[(size_t)e * 16 + k];
    }
    const v8bf* xlp = (const v8bf*)(xl + (size_t)src * 256 + h * 64);
    const v8bf* xrp = (const v8bf*)(xr + (size_t)dst * 256 + h * 64);
    const float* wrow = we_s + h * 64 * 16;
    const float* arow = att_s + h * 64;
    const float* eps  = ep_self + h * 64;

    float acc = 0.f;
#pragma unroll
    for (int ch = 0; ch < 8; ++ch) {          // 8 chunks of 8 channels: 16B vector loads
        v8bf xlc = xlp[ch];
        v8bf xrc = xrp[ch];
#pragma unroll
        for (int i = 0; i < 8; ++i) {
            int c = ch * 8 + i;
            float ep;
            if (real) {
                ep = 0.f;
                const float* w = wrow + c * 16;
#pragma unroll
                for (int k = 0; k < 16; ++k) ep += rel[k] * w[k];
            } else {
                ep = eps[c];
            }
            float m = (float)xlc[i] + (float)xrc[i] + ep;
            m = m > 0.f ? m : 0.2f * m;       // leaky_relu(0.2)
            acc += m * arow[c];
        }
    }
    logits[(size_t)e * 4 + h] = acc;
    atomicMax(mx + (size_t)dst * 4 + h, fenc(acc));
}

// ---------------- Pass 2: exp(l - max) + segment sum ----------------
__global__ void edge_exp_kernel(const float* __restrict__ logits, const unsigned* __restrict__ mx,
                                const int* __restrict__ edges, int E, int Et,
                                float* __restrict__ exb, float* __restrict__ den) {
    int gid = blockIdx.x * blockDim.x + threadIdx.x;
    if (gid >= Et * 4) return;
    int e = gid >> 2, h = gid & 3;
    int dst = (e < E) ? edges[E + e] : (e - E);
    float l = logits[gid];
    float m = fdec(mx[(size_t)dst * 4 + h]);
    float ex = __expf(l - m);
    exb[gid] = ex;
    unsafeAtomicAdd(den + (size_t)dst * 4 + h, ex);
}

// ---------------- Pass 3: alpha-weighted scatter of xl[src] ----------------
__global__ void edge_scatter_kernel(const float* __restrict__ exb, const float* __restrict__ den,
                                    const __bf16* __restrict__ xl,
                                    const int* __restrict__ edges, int E, int Et,
                                    float* __restrict__ outacc) {
    int gid = blockIdx.x * blockDim.x + threadIdx.x;
    if (gid >= Et * 4) return;
    int e = gid >> 2, h = gid & 3;
    int src, dst;
    if (e < E) { src = edges[e]; dst = edges[E + e]; }
    else       { src = dst = e - E; }
    float alpha = exb[gid] / den[(size_t)dst * 4 + h];
    const v8bf* xlp = (const v8bf*)(xl + (size_t)src * 256 + h * 64);
    float* op = outacc + (size_t)dst * 256 + h * 64;
#pragma unroll
    for (int ch = 0; ch < 8; ++ch) {
        v8bf xlc = xlp[ch];
#pragma unroll
        for (int i = 0; i < 8; ++i)
            unsafeAtomicAdd(op + ch * 8 + i, (float)xlc[i] * alpha);
    }
}

// ---------------- finalize: head-mean + bias + residual + relu ----------------
__global__ void finalize_kernel(const float* __restrict__ outacc, const float* __restrict__ gat_b,
                                float* __restrict__ hf, __bf16* __restrict__ hb,
                                float* __restrict__ out_final, int N) {
    int gid = blockIdx.x * blockDim.x + threadIdx.x;
    if (gid >= N * 64) return;
    int n = gid >> 6, c = gid & 63;
    const float* op = outacc + (size_t)n * 256 + c;
    float s = 0.25f * (op[0] + op[64] + op[128] + op[192]) + gat_b[c] + hf[gid];
    float r = s > 0.f ? s : 0.f;
    hf[gid] = r;
    hb[gid] = (__bf16)r;
    if (out_final) out_final[gid] = r;
}

// ---------------- host launch ----------------
static inline int cdiv(long long a, long long b) { return (int)((a + b - 1) / b); }

extern "C" void kernel_launch(void* const* d_in, const int* in_sizes, int n_in,
                              void* d_out, int out_size, void* d_ws, size_t ws_size,
                              hipStream_t stream) {
    const float* nodes    = (const float*)d_in[0];
    const int*   edges    = (const int*)d_in[1];
    const float* relation = (const float*)d_in[2];
    const float* temb     = (const float*)d_in[3];
    const float* tyemb    = (const float*)d_in[4];
    const float* bwemb    = (const float*)d_in[5];
    const float* ilemb    = (const float*)d_in[6];
    const float* proj_w   = (const float*)d_in[7];
    const float* proj_b   = (const float*)d_in[8];
    const float* lin_l_w  = (const float*)d_in[9];
    const float* lin_l_b  = (const float*)d_in[10];
    const float* lin_r_w  = (const float*)d_in[11];
    const float* lin_r_b  = (const float*)d_in[12];
    const float* lin_e_w  = (const float*)d_in[13];
    const float* att      = (const float*)d_in[14];
    const float* gat_b    = (const float*)d_in[15];

    const int N  = in_sizes[0] / 20;
    const int E  = in_sizes[1] / 2;
    const int Et = E + N;

    // workspace carve-out (256B aligned)
    char* p = (char*)d_ws;
    auto alloc = [&](size_t bytes) -> void* {
        void* r = (void*)p;
        p += (bytes + 255) & ~(size_t)255;
        return r;
    };
    __bf16* h0b    = (__bf16*)alloc((size_t)N * 64 * 2);
    float*  hf     = (float*) alloc((size_t)N * 64 * 4);
    __bf16* hb     = (__bf16*)alloc((size_t)N * 64 * 2);
    __bf16* xlb    = (__bf16*)alloc((size_t)N * 256 * 2);
    __bf16* xrb    = (__bf16*)alloc((size_t)N * 256 * 2);
    float*  logits = (float*) alloc((size_t)Et * 4 * 4);
    float*  exb    = (float*) alloc((size_t)Et * 4 * 4);
    unsigned* mx   = (unsigned*)alloc((size_t)N * 4 * 4);
    float*  den    = (float*) alloc((size_t)N * 4 * 4);
    float*  outacc = (float*) alloc((size_t)N * 256 * 4);
    float*  relsum = (float*) alloc(64);
    float*  epself = (float*) alloc(256 * 4);
    __bf16* wpack4  = (__bf16*)alloc((size_t)1024 * 4 * 2);    // proj weights, packed
    __bf16* wpack16 = (__bf16*)alloc((size_t)1024 * 16 * 2);   // per-GEMM packed 256x64 weights

    const int TB = 256;

    zero16_kernel<<<1, 32, 0, stream>>>(relsum);
    embed_kernel<<<cdiv((long long)N * 64, TB), TB, 0, stream>>>(nodes, temb, tyemb, bwemb, ilemb, h0b, N);
    relsum_kernel<<<256, TB, 0, stream>>>(relation, E, relsum);

    // h = h0 @ proj_w^T + proj_b   (f32 + bf16 copies)
    pack_w_kernel<4><<<cdiv(1024 * 4, TB), TB, 0, stream>>>(proj_w, wpack4);
    gemm_wmma_kernel<4><<<cdiv(N, 512), TB, 0, stream>>>(h0b, wpack4, proj_b, hf, hb, N);

    for (int l = 0; l < 2; ++l) {
        const float* wl = lin_l_w + (size_t)l * 256 * 64;
        const float* bl = lin_l_b + (size_t)l * 256;
        const float* wr = lin_r_w + (size_t)l * 256 * 64;
        const float* br = lin_r_b + (size_t)l * 256;
        const float* we = lin_e_w + (size_t)l * 256 * 16;
        const float* al = att + (size_t)l * 256;
        const float* gb = gat_b + (size_t)l * 64;

        epself_kernel<<<1, 256, 0, stream>>>(relsum, 1.0f / (float)E, we, epself);

        pack_w_kernel<16><<<cdiv(1024 * 16, TB), TB, 0, stream>>>(wl, wpack16);
        gemm_wmma_kernel<16><<<cdiv(N, 512), TB, 0, stream>>>(hb, wpack16, bl, (float*)nullptr, xlb, N);
        pack_w_kernel<16><<<cdiv(1024 * 16, TB), TB, 0, stream>>>(wr, wpack16);
        gemm_wmma_kernel<16><<<cdiv(N, 512), TB, 0, stream>>>(hb, wpack16, br, (float*)nullptr, xrb, N);

        layer_init_kernel<<<cdiv((long long)N * 256, TB), TB, 0, stream>>>(outacc, mx, den, N);

        edge_logits_kernel<<<cdiv((long long)Et * 4, TB), TB, 0, stream>>>(
            xlb, xrb, relation, we, al, epself, edges, E, Et, logits, mx);
        edge_exp_kernel<<<cdiv((long long)Et * 4, TB), TB, 0, stream>>>(
            logits, mx, edges, E, Et, exb, den);
        edge_scatter_kernel<<<cdiv((long long)Et * 4, TB), TB, 0, stream>>>(
            exb, den, xlb, edges, E, Et, outacc);

        finalize_kernel<<<cdiv((long long)N * 64, TB), TB, 0, stream>>>(
            outacc, gb, hf, hb, (l == 1) ? (float*)d_out : (float*)nullptr, N);
    }
}
